// CausalSelfAttention_85676007621287
// MI455X (gfx1250) — matrix operand order repaired and checked by reading, not compiled, not run on hardware
//
#include <hip/hip_runtime.h>
#include <hip/hip_bf16.h>

// ---------------------------------------------------------------------------
// CDNA5 (gfx1250) fused causal self-attention, f16 WMMA everywhere.
// B=2, T=2048, C=1024, NH=16, hd=64.
// - LDS fragment gathers are 16-contiguous-half reads -> 2x ds_load_b128
// - B-fragments preloaded so the 4 v_wmma issue back-to-back per step
// - global->LDS staging via GLOBAL_LOAD_ASYNC_TO_LDS_B128 (ASYNCcnt),
//   double-buffered in BOTH the GEMM and the attention k-loop.
// ---------------------------------------------------------------------------

typedef __attribute__((ext_vector_type(16))) _Float16 v16h;
typedef __attribute__((ext_vector_type(8)))  float    v8f;

#define EMB   1024
#define NHEAD 16
#define HD    64
#define TDIM  2048
#define BDIM  2
#define MROWS (BDIM * TDIM)   // 4096 token rows

#define APAD 40   // padded row stride (halfs) for 32-wide K tiles
#define QPAD 72   // padded row stride (halfs) for 64-wide tiles

// ---------------- async global->LDS copy helpers ---------------------------

// Copy 16 f16 (32B) per lane: global (base + elem_off) -> LDS dst.
// 2x GLOBAL_LOAD_ASYNC_TO_LDS_B128, GVS mode; INST_OFFSET applies to both
// the global and LDS addresses (ISA async pseudocode).
__device__ __forceinline__ void async_copy16(const _Float16* gbase,
                                             uint32_t elem_off,
                                             _Float16* lds_dst) {
  const uint32_t ldsa = (uint32_t)(uintptr_t)lds_dst;   // low 32b = LDS offset
  const uint32_t gofs = elem_off * 2u;                  // byte offset
  const unsigned long long gb = (unsigned long long)(uintptr_t)gbase;
  asm volatile("global_load_async_to_lds_b128 %0, %1, %2 offset:0\n\t"
               "global_load_async_to_lds_b128 %0, %1, %2 offset:16"
               :: "v"(ldsa), "v"(gofs), "s"(gb) : "memory");
}

__device__ __forceinline__ void wait_async_le4() {
  asm volatile("s_wait_asynccnt 0x4" ::: "memory");
}
__device__ __forceinline__ void wait_async_0() {
  asm volatile("s_wait_asynccnt 0x0" ::: "memory");
}

// ---------------- fragment gathers (k = 16*hi + j in both A and B) ----------

// A-matrix 16x32 f16, tile row-major (M x K), row stride `stride` (halfs).
__device__ __forceinline__ v16h load_a_frag(const _Float16* tile, int stride) {
  const int lane = threadIdx.x & 31;
  const int m  = lane & 15;
  const int hi = lane >> 4;
  v16h a;
#pragma unroll
  for (int j = 0; j < 16; ++j) {
    const int k = hi * 16 + j;
    a[j] = tile[m * stride + k];
  }
  return a;
}

// B-matrix 32x16 (K x N) given a TRANSPOSED tile (N x K row-major).
__device__ __forceinline__ v16h load_bT_frag(const _Float16* tile, int stride) {
  const int lane = threadIdx.x & 31;
  const int n  = lane & 15;
  const int kb = (lane >> 4) * 16;
  v16h b;
#pragma unroll
  for (int j = 0; j < 16; ++j) b[j] = tile[n * stride + kb + j];
  return b;
}

// ---------------- fp32 -> f16 conversion kernels ---------------------------

__global__ void cvt_f32_f16(const float* __restrict__ in,
                            _Float16* __restrict__ out, int n) {
  int i = blockIdx.x * blockDim.x + threadIdx.x;
  if (i < n) out[i] = (_Float16)in[i];
}

// W [K,N] row-major (fp32) -> WT [N,K] row-major (f16); LDS-tiled transpose.
__global__ __launch_bounds__(256) void cvt_transpose_f16(
    const float* __restrict__ in, _Float16* __restrict__ out, int K, int N) {
  __shared__ float tile[32][33];
  const int bx = blockIdx.x * 32;  // N offset
  const int by = blockIdx.y * 32;  // K offset
  const int tx = threadIdx.x, ty = threadIdx.y;  // (32, 8)
#pragma unroll
  for (int r = 0; r < 32; r += 8)
    tile[ty + r][tx] = in[(size_t)(by + ty + r) * N + bx + tx];
  __syncthreads();
#pragma unroll
  for (int r = 0; r < 32; r += 8)
    out[(size_t)(bx + ty + r) * K + by + tx] = (_Float16)tile[tx][ty + r];
}

// ---------------- generic 64x64-tile WMMA GEMM ------------------------------
// C[M,N] = A[M,K] @ W[K,N] + bias[N], W supplied TRANSPOSED: WT [N,K].
// mode 0: f16 out, [B,NH,T,hd] layout, per-element scale   (Q,K projections)
// mode 1: f32 out, row-major [M,N]                          (output proj)
// mode 2: f16 out, [B,NH,hd,T] layout (transposed V)        (V projection)

__global__ __launch_bounds__(128) void gemm_wmma(
    const _Float16* __restrict__ A, const _Float16* __restrict__ WT,
    const float* __restrict__ bias, void* __restrict__ outp,
    int M, int K, int N, int mode, float scale) {
  __shared__ alignas(16) _Float16 As [2][64 * APAD];
  __shared__ alignas(16) _Float16 Bsw[2][64 * APAD];

  const int tid  = threadIdx.x;
  const int w    = tid >> 5;
  const int lane = tid & 31;
  const int n    = lane & 15;
  const int hi   = lane >> 4;
  const int nblk = blockIdx.x * 64;
  const int mblk = blockIdx.y * 64;

  // per-thread staging slot: row si (0..63), 16-half chunk sp (0..1)
  const int si = tid >> 1, sp = tid & 1;

  v8f acc[4] = {};

  // prologue: stage tile 0
  async_copy16(A,  (uint32_t)((mblk + si) * K + 0 + sp * 16),
               &As[0][si * APAD + sp * 16]);
  async_copy16(WT, (uint32_t)((nblk + si) * K + 0 + sp * 16),
               &Bsw[0][si * APAD + sp * 16]);

  int cur = 0;
  for (int k0 = 0; k0 < K; k0 += 32) {
    if (k0 + 32 < K) {
      async_copy16(A,  (uint32_t)((mblk + si) * K + k0 + 32 + sp * 16),
                   &As[cur ^ 1][si * APAD + sp * 16]);
      async_copy16(WT, (uint32_t)((nblk + si) * K + k0 + 32 + sp * 16),
                   &Bsw[cur ^ 1][si * APAD + sp * 16]);
      wait_async_le4();   // in-order completion: tile k0 has landed
    } else {
      wait_async_0();
    }
    __syncthreads();

    // preload A + all 4 B fragments, then 4 back-to-back WMMAs
    const v16h a = load_a_frag(&As[cur][(w * 16) * APAD], APAD);
    v16h bfr[4];
#pragma unroll
    for (int t = 0; t < 4; ++t)
      bfr[t] = load_bT_frag(&Bsw[cur][(t * 16) * APAD], APAD);
#pragma unroll
    for (int t = 0; t < 4; ++t)
      acc[t] = __builtin_amdgcn_wmma_f32_16x16x32_f16(
          false, a, false, bfr[t], (short)0, acc[t], false, false);

    __syncthreads();   // all waves done reading before buffer is re-staged
    cur ^= 1;
  }

  // epilogue
#pragma unroll
  for (int t = 0; t < 4; ++t) {
#pragma unroll
    for (int r = 0; r < 8; ++r) {
      const int row = mblk + w * 16 + r + 8 * hi;
      const int col = nblk + t * 16 + n;
      const float v = acc[t][r] + bias[col];
      if (mode == 1) {
        ((float*)outp)[(size_t)row * N + col] = v;
      } else {
        const int bb = row / TDIM, tt = row % TDIM;
        const int h = col >> 6, d = col & 63;
        const size_t bh = (size_t)bb * NHEAD + h;
        const size_t idx = (mode == 0) ? (bh * TDIM + tt) * HD + d    // [B,H,T,hd]
                                       : (bh * HD + d) * TDIM + tt;   // [B,H,hd,T]
        ((_Float16*)outp)[idx] = (_Float16)(v * scale);
      }
    }
  }
}

// ---------------- flash attention (causal, online softmax) -----------------
// grid: (T/64, NH, B); block: 128 threads = 4 waves, wave w owns 16 queries.
// q,k in [B,H,T,hd]; v in [B,H,hd,T]. K/V staging double-buffered async.

__global__ __launch_bounds__(128) void attn_wmma(
    const _Float16* __restrict__ qh, const _Float16* __restrict__ kh,
    const _Float16* __restrict__ vh, _Float16* __restrict__ yh) {
  __shared__ alignas(16) _Float16 qs [64 * QPAD];      // 64 q rows x 64 hd
  __shared__ alignas(16) _Float16 ks [2][32 * QPAD];   // 32 keys  x 64 hd
  __shared__ alignas(16) _Float16 vsT[2][64 * APAD];   // 64 hd    x 32 keys
  __shared__ alignas(16) _Float16 ps [4][16 * APAD];   // per-wave P bounce

  const int tid  = threadIdx.x;
  const int w    = tid >> 5;
  const int lane = tid & 31;
  const int n    = lane & 15;
  const int hi   = lane >> 4;
  const int q0   = blockIdx.x * 64;
  const int h    = blockIdx.y;
  const int b    = blockIdx.z;
  const size_t base  = ((size_t)b * NHEAD + h) * TDIM * HD;   // q,k base
  const size_t basev = ((size_t)b * NHEAD + h) * HD * TDIM;   // v base (hd,T)

  // stage Q block 64x64 async: row i=tid/2, 32-half chunk p=tid&1
  {
    const int i = tid >> 1, p = tid & 1;
    const uint32_t off = (uint32_t)(base + (size_t)(q0 + i) * HD + p * 32);
    async_copy16(qh, off,      qs + i * QPAD + p * 32);
    async_copy16(qh, off + 16, qs + i * QPAD + p * 32 + 16);
  }
  wait_async_0();
  __syncthreads();
  const v16h qa0 = load_a_frag(qs + (w * 16) * QPAD + 0,  QPAD);
  const v16h qa1 = load_a_frag(qs + (w * 16) * QPAD + 32, QPAD);

  // per-thread K/V staging slots
  const int kk = tid >> 2, p3 = tid & 3;   // K: row kk, chunk p3
  const int dv = tid >> 1, p1 = tid & 1;   // V: row dv, chunk p1
  auto issue_kv = [&](int buf, int kst) {
    async_copy16(kh, (uint32_t)(base + (size_t)(kst + kk) * HD + p3 * 16),
                 &ks[buf][kk * QPAD + p3 * 16]);
    async_copy16(vh, (uint32_t)(basev + (size_t)dv * TDIM + kst + p1 * 16),
                 &vsT[buf][dv * APAD + p1 * 16]);
  };

  v8f acc[4] = {};
  float mrow[8], lrow[8];
#pragma unroll
  for (int r = 0; r < 8; ++r) { mrow[r] = -1e30f; lrow[r] = 0.0f; }

  const int qmax_w = q0 + w * 16 + 15;
  const int kend   = q0 + 64;

  issue_kv(0, 0);   // prologue
  int cur = 0;
  for (int kst = 0; kst < kend; kst += 32) {
    if (kst + 32 < kend) {
      issue_kv(cur ^ 1, kst + 32);   // 4 async instrs per thread
      wait_async_le4();              // previous 4 (tile kst) have landed
    } else {
      wait_async_0();
    }
    __syncthreads();

    if (kst <= qmax_w) {  // wave-uniform: EXEC stays all-1 for WMMA
      // ---- S = Q @ K^T : preload 4 K^T frags, 4 back-to-back WMMAs ----
      const _Float16* kcur = &ks[cur][0];
      v16h bk[4];
      bk[0] = load_bT_frag(kcur + 0  * QPAD + 0,  QPAD);
      bk[1] = load_bT_frag(kcur + 16 * QPAD + 0,  QPAD);
      bk[2] = load_bT_frag(kcur + 0  * QPAD + 32, QPAD);
      bk[3] = load_bT_frag(kcur + 16 * QPAD + 32, QPAD);
      v8f S0 = {}, S1 = {};
      S0 = __builtin_amdgcn_wmma_f32_16x16x32_f16(
          false, qa0, false, bk[0], (short)0, S0, false, false);
      S1 = __builtin_amdgcn_wmma_f32_16x16x32_f16(
          false, qa0, false, bk[1], (short)0, S1, false, false);
      S0 = __builtin_amdgcn_wmma_f32_16x16x32_f16(
          false, qa1, false, bk[2], (short)0, S0, false, false);
      S1 = __builtin_amdgcn_wmma_f32_16x16x32_f16(
          false, qa1, false, bk[3], (short)0, S1, false, false);

      // ---- causal mask + online softmax (rows striped per C-layout) ----
      float p0[8], p1v[8];
#pragma unroll
      for (int r = 0; r < 8; ++r) {
        const int qidx = q0 + w * 16 + r + 8 * hi;
        float s0 = (kst + n      <= qidx) ? S0[r] : -1e30f;
        float s1 = (kst + 16 + n <= qidx) ? S1[r] : -1e30f;
        float t = fmaxf(s0, s1);
#pragma unroll
        for (int off = 1; off < 16; off <<= 1) t = fmaxf(t, __shfl_xor(t, off));
        const float mn = fmaxf(mrow[r], t);
        const float sc = __expf(mrow[r] - mn);
        const float e0 = __expf(s0 - mn);
        const float e1 = __expf(s1 - mn);
        float rs = e0 + e1;
#pragma unroll
        for (int off = 1; off < 16; off <<= 1) rs += __shfl_xor(rs, off);
        lrow[r] = lrow[r] * sc + rs;
        mrow[r] = mn;
#pragma unroll
        for (int t2 = 0; t2 < 4; ++t2) acc[t2][r] *= sc;
        p0[r] = e0; p1v[r] = e1;
      }

      // ---- re-layout P (C-format) -> A-format via per-wave LDS bounce ----
      _Float16* pw = &ps[w][0];
#pragma unroll
      for (int r = 0; r < 8; ++r) {
        const int row = r + 8 * hi;
        pw[row * APAD + n]      = (_Float16)p0[r];
        pw[row * APAD + 16 + n] = (_Float16)p1v[r];
      }
      const v16h pa = load_a_frag(pw, APAD);  // DS ops in-order within wave

      // ---- acc += P @ V : preload 4 V frags, 4 back-to-back WMMAs ----
      const _Float16* vcur = &vsT[cur][0];
      v16h bv[4];
#pragma unroll
      for (int t = 0; t < 4; ++t)
        bv[t] = load_bT_frag(vcur + (t * 16) * APAD, APAD);
#pragma unroll
      for (int t = 0; t < 4; ++t)
        acc[t] = __builtin_amdgcn_wmma_f32_16x16x32_f16(
            false, pa, false, bv[t], (short)0, acc[t], false, false);
    }
    __syncthreads();   // all waves done with buffers before restage
    cur ^= 1;
  }

  // ---- normalize and store y in [B*T, C] f16 (token row, col = h*64+d) ----
#pragma unroll
  for (int r = 0; r < 8; ++r) {
    const float inv = 1.0f / lrow[r];
    const int row = q0 + w * 16 + r + 8 * hi;
    const size_t tok = (size_t)b * TDIM + row;
#pragma unroll
    for (int t = 0; t < 4; ++t)
      yh[tok * EMB + h * 64 + t * 16 + n] = (_Float16)(acc[t][r] * inv);
  }
}

// ---------------------------------------------------------------------------

extern "C" void kernel_launch(void* const* d_in, const int* in_sizes, int n_in,
                              void* d_out, int out_size, void* d_ws, size_t ws_size,
                              hipStream_t stream) {
  const float* x  = (const float*)d_in[0];
  const float* Wq = (const float*)d_in[1];
  const float* bq = (const float*)d_in[2];
  const float* Wk = (const float*)d_in[3];
  const float* bk = (const float*)d_in[4];
  const float* Wv = (const float*)d_in[5];
  const float* bv = (const float*)d_in[6];
  const float* Wo = (const float*)d_in[7];
  const float* bo = (const float*)d_in[8];

  _Float16* ws = (_Float16*)d_ws;
  const size_t NX = (size_t)MROWS * EMB;      // 4,194,304
  const size_t NW = (size_t)EMB * EMB;        // 1,048,576
  _Float16* xh  = ws;
  _Float16* Wqh = xh  + NX;   // all weights stored TRANSPOSED [out,in]
  _Float16* Wkh = Wqh + NW;
  _Float16* Wvh = Wkh + NW;
  _Float16* Woh = Wvh + NW;
  _Float16* qh  = Woh + NW;   // [B,H,T,hd]
  _Float16* kh  = qh  + NX;   // [B,H,T,hd]
  _Float16* vh  = kh  + NX;   // [B,H,hd,T]
  _Float16* yh  = vh  + NX;   // [B*T, C]

  // fp32 -> f16 staging (weights transposed for contiguous B-fragments)
  cvt_f32_f16<<<(int)((NX + 255) / 256), 256, 0, stream>>>(x, xh, (int)NX);
  dim3 tb(32, 8), tg(EMB / 32, EMB / 32);
  cvt_transpose_f16<<<tg, tb, 0, stream>>>(Wq, Wqh, EMB, EMB);
  cvt_transpose_f16<<<tg, tb, 0, stream>>>(Wk, Wkh, EMB, EMB);
  cvt_transpose_f16<<<tg, tb, 0, stream>>>(Wv, Wvh, EMB, EMB);
  cvt_transpose_f16<<<tg, tb, 0, stream>>>(Wo, Woh, EMB, EMB);

  // Q/K/V projections (Q pre-scaled by 1/sqrt(hd) = 0.125)
  dim3 gg(EMB / 64, MROWS / 64);
  gemm_wmma<<<gg, 128, 0, stream>>>(xh, Wqh, bq, qh, MROWS, EMB, EMB, 0, 0.125f);
  gemm_wmma<<<gg, 128, 0, stream>>>(xh, Wkh, bk, kh, MROWS, EMB, EMB, 0, 1.0f);
  gemm_wmma<<<gg, 128, 0, stream>>>(xh, Wvh, bv, vh, MROWS, EMB, EMB, 2, 1.0f);

  // causal flash attention
  dim3 ga(TDIM / 64, NHEAD, BDIM);
  attn_wmma<<<ga, 128, 0, stream>>>(qh, kh, vh, yh);

  // output projection -> fp32 d_out
  gemm_wmma<<<gg, 128, 0, stream>>>(yh, Woh, bo, d_out, MROWS, EMB, EMB, 1, 1.0f);
}
